// SwinTransformerBlock_27479200760376
// MI455X (gfx1250) — matrix operand order repaired and checked by reading, not compiled, not run on hardware
//
#include <hip/hip_runtime.h>
#include <hip/hip_bf16.h>
#include <math.h>

typedef __attribute__((ext_vector_type(16))) _Float16 v16h;
typedef __attribute__((ext_vector_type(8)))  float    v8f;
typedef unsigned int u32x4 __attribute__((ext_vector_type(4)));
typedef int          i32x8 __attribute__((ext_vector_type(8)));
typedef int          i32x4 __attribute__((ext_vector_type(4)));

#define WMMA_F16(a, b, c) \
  __builtin_amdgcn_wmma_f32_16x16x32_f16(false, (a), false, (b), (short)0, (c), false, false)

// ---------------------------------------------------------------------------
// geometry: B=4, H=W=64, C=512, WS=16, SHIFT=8, NH=16, hd=32
// token order t = win*256 + i ; win = b*16 + wh*4 + ww ; i = r*16 + c
// ---------------------------------------------------------------------------
__device__ __forceinline__ long canvas_base(int t) {
  int i = t & 255, win = t >> 8;
  int b = win >> 4, wimg = win & 15, wh = wimg >> 2, ww = wimg & 3;
  int r = i >> 4, cq = i & 15;
  int gh = (wh * 16 + r + 8) & 63, gw = (ww * 16 + cq + 8) & 63;
  return ((long)b * 4096 + gh * 64 + gw) * 512;
}
__device__ __forceinline__ int region_of(int g) { return g < 48 ? 0 : (g < 56 ? 1 : 2); }

// ---------------------------------------------------------------------------
__global__ __launch_bounds__(256) void cvt_f16(const float* __restrict__ s,
                                               _Float16* __restrict__ d, int n) {
  int i = blockIdx.x * 256 + threadIdx.x;
  if (i < n) d[i] = (_Float16)s[i];
}

__global__ __launch_bounds__(256) void gather_win(const float* __restrict__ x,
                                                  _Float16* __restrict__ xw) {
  long e = (long)blockIdx.x * 256 + threadIdx.x;  // 16384*512 elems
  int c = (int)(e & 511);
  int t = (int)(e >> 9);
  xw[e] = (_Float16)x[canvas_base(t) + c];
}

// ---------------------------------------------------------------------------
// CPB MLP: bias_tbl[t][h] for t in 0..960 (31x31 table), h in 0..15
// ---------------------------------------------------------------------------
__global__ __launch_bounds__(128) void cpb_kernel(const float* __restrict__ w1,
                                                  const float* __restrict__ b1,
                                                  const float* __restrict__ w2,
                                                  float* __restrict__ btbl) {
  __shared__ float hid[512];
  int t = blockIdx.x;
  int i = t / 31, j = t % 31;
  float rr = (float)(i - 15) * (8.0f / 15.0f);
  float cc = (float)(j - 15) * (8.0f / 15.0f);
  float v0 = (float)((rr > 0.f) - (rr < 0.f)) * log2f(fabsf(rr) + 1.0f) * (1.0f / 3.0f);
  float v1 = (float)((cc > 0.f) - (cc < 0.f)) * log2f(fabsf(cc) + 1.0f) * (1.0f / 3.0f);
  for (int h = threadIdx.x; h < 512; h += 128) {
    float z = v0 * w1[h * 2] + v1 * w1[h * 2 + 1] + b1[h];
    hid[h] = fmaxf(z, 0.0f);
  }
  __syncthreads();
  if (threadIdx.x < 16) {
    int hh = threadIdx.x;
    float s = 0.f;
    for (int u = 0; u < 512; ++u) s += hid[u] * w2[hh * 512 + u];
    btbl[t * 16 + hh] = s;
  }
}

// ---------------------------------------------------------------------------
// f16 WMMA GEMM: out[M,N] = A[M,K] * W[N,K]^T (+bias epilogues)
// wave = 16Mx64N tile (4 accumulators, A-frag reused 4x); block = 4 waves = 256N
//  mode 0: qkv scatter (+q_bias / 0 / v_bias) -> qkvout f32 [3][64][16][256][32]
//  mode 1: +bias -> of32 row-major
//  mode 2: +bias, exact GELU -> of16 row-major
// ---------------------------------------------------------------------------
__global__ __launch_bounds__(128) void gemm_f16(
    const _Float16* __restrict__ A, const _Float16* __restrict__ W,
    int M, int N, int K, const float* __restrict__ bias, int mode,
    float* __restrict__ of32, _Float16* __restrict__ of16,
    const float* __restrict__ qb, const float* __restrict__ vb,
    float* __restrict__ qkvout) {
  int lane = threadIdx.x & 31, wv = threadIdx.x >> 5;
  int half = lane >> 4, ln = lane & 15;
  int mtile = blockIdx.x;
  int nquad = blockIdx.y * 4 + wv;  // wave's 64-wide N slab, in units of 64
  if (nquad * 64 >= N || mtile * 16 >= M) return;

  const _Float16* arow = A + (long)(mtile * 16 + ln) * K;  // A lane row: M = ln
  const _Float16* wrow0 = W + (long)(nquad * 64 + ln) * K; // B lane cols
  const _Float16* wrow1 = wrow0 + (long)16 * K;
  const _Float16* wrow2 = wrow0 + (long)32 * K;
  const _Float16* wrow3 = wrow0 + (long)48 * K;

  v8f acc[4] = {};
  for (int k0 = 0; k0 < K; k0 += 32) {
    v16h a, b0, b1, b2, b3;
#pragma unroll
    for (int v = 0; v < 8; ++v) {
      int ka = k0 + 2 * v + ((v >= 4) ? 8 : 0) + 8 * half;  // A k-pair pattern
      a[2 * v] = arow[ka];
      a[2 * v + 1] = arow[ka + 1];
      int kb = k0 + 2 * v + ((v >= 4) ? 8 : 0) + 8 * half;  // B k-pair pattern
      b0[2 * v] = wrow0[kb]; b0[2 * v + 1] = wrow0[kb + 1];
      b1[2 * v] = wrow1[kb]; b1[2 * v + 1] = wrow1[kb + 1];
      b2[2 * v] = wrow2[kb]; b2[2 * v + 1] = wrow2[kb + 1];
      b3[2 * v] = wrow3[kb]; b3[2 * v + 1] = wrow3[kb + 1];
    }
    acc[0] = WMMA_F16(a, b0, acc[0]);
    acc[1] = WMMA_F16(a, b1, acc[1]);
    acc[2] = WMMA_F16(a, b2, acc[2]);
    acc[3] = WMMA_F16(a, b3, acc[3]);
  }
#pragma unroll
  for (int j = 0; j < 4; ++j) {
#pragma unroll
    for (int r = 0; r < 8; ++r) {
      int m = mtile * 16 + r + 8 * half;
      int n = nquad * 64 + j * 16 + ln;
      float val = acc[j][r];
      if (mode == 0) {
        float bb = (n < 512) ? qb[n] : ((n < 1024) ? 0.0f : vb[n - 1024]);
        val += bb;
        int which = n >> 9, rem = n & 511, head = rem >> 5, d = rem & 31;
        int win = m >> 8, i = m & 255;
        qkvout[((((long)which * 64 + win) * 16 + head) * 256 + i) * 32 + d] = val;
      } else if (mode == 1) {
        of32[(long)m * N + n] = val + bias[n];
      } else {
        float xg = val + bias[n];
        float g = 0.5f * xg * (1.0f + erff(xg * 0.70710678118f));
        of16[(long)m * N + n] = (_Float16)g;
      }
    }
  }
}

// ---------------------------------------------------------------------------
// cosine-attn prep: qn = q/||q|| * exp(min(ls,log100)), kn = k/||k||, v -> f16
// ---------------------------------------------------------------------------
__global__ __launch_bounds__(256) void qkv_norm(const float* __restrict__ qkv,
                                                const float* __restrict__ ls,
                                                _Float16* __restrict__ qn,
                                                _Float16* __restrict__ kn,
                                                _Float16* __restrict__ vh) {
  int lane = threadIdx.x & 31, wv = threadIdx.x >> 5;
  long rowid = (long)blockIdx.x * 8 + wv;  // 262144 rows = 64*16*256
  int head = (int)((rowid >> 8) & 15);
  const long stride = (long)262144 * 32;
  long off = rowid * 32 + lane;
  float q = qkv[off], k = qkv[stride + off], v = qkv[2 * stride + off];
  float sq = q * q, sk = k * k;
#pragma unroll
  for (int o = 1; o < 32; o <<= 1) {
    sq += __shfl_xor(sq, o, 32);
    sk += __shfl_xor(sk, o, 32);
  }
  float scale = expf(fminf(ls[head], 4.6051702f));
  qn[off] = (_Float16)(q / fmaxf(sqrtf(sq), 1e-12f) * scale);
  kn[off] = (_Float16)(k / fmaxf(sqrtf(sk), 1e-12f));
  vh[off] = (_Float16)v;
}

// ---------------------------------------------------------------------------
// fused attention: block = (win, head, quad); 4 waves, each one 16-row tile.
// kn tile (256x32 f16 = 16KB contiguous) staged to LDS via the Tensor Data
// Mover (TDM) when available, else a manual copy loop.
// ---------------------------------------------------------------------------
__global__ __launch_bounds__(128) void attn_kernel(const _Float16* __restrict__ qn,
                                                   const _Float16* __restrict__ kn,
                                                   const _Float16* __restrict__ vt,
                                                   const float* __restrict__ btbl,
                                                   _Float16* __restrict__ aout) {
  __shared__ _Float16 lds_kn[256 * 32];      // 16 KB
  __shared__ _Float16 lds_p[4][16 * 256];    // 32 KB
  int bid = blockIdx.x;                       // 64*16*4 blocks
  int win = bid >> 6, rem = bid & 63, head = rem >> 2, quad = rem & 3;
  int tid = threadIdx.x, lane = tid & 31, wv = tid >> 5;
  int half = lane >> 4, ln = lane & 15;
  long base = (long)(win * 16 + head) * 256 * 32;

#if defined(__has_builtin) && __has_builtin(__builtin_amdgcn_tensor_load_to_lds) && \
    __has_builtin(__builtin_amdgcn_s_wait_tensorcnt)
  if (tid < 32) {  // wave 0 issues one TDM descriptor: 2048 x 8B contiguous tile
    unsigned long long gaddr = (unsigned long long)(const void*)(kn + base);
    unsigned int lds_off = (unsigned int)(unsigned long long)(void*)&lds_kn[0];
    u32x4 g0;
    g0[0] = 1u;                                          // count=1 (valid D#)
    g0[1] = lds_off;                                     // lds_addr
    g0[2] = (unsigned int)gaddr;                         // global_addr[31:0]
    g0[3] = (unsigned int)(gaddr >> 32) | (2u << 30);    // addr[56:32] | type=2
    i32x8 g1;
    g1[0] = 3 << 16;                 // data_size = 8B; no multicast/pad/iterate
    g1[1] = (int)(2048u << 16);      // tensor_dim0[15:0] = 2048
    g1[2] = (int)(1u << 16);         // tensor_dim0 hi = 0 ; tensor_dim1 = 1
    g1[3] = (int)(2048u << 16);      // tile_dim0 = 2048
    g1[4] = 1;                       // tile_dim1 = 1
    g1[5] = 2048;                    // tensor_dim0_stride = 2048
    g1[6] = (int)(2048u << 16);      // tensor_dim1_stride[15:0] = 2048
    g1[7] = 0;
    i32x4 z4 = {};
    i32x8 z8 = {};
    // 6-arg toolchain variant: (g0, g1, g2, g3, g4, cpol)
    __builtin_amdgcn_tensor_load_to_lds(g0, g1, z4, z4, z8, 0);
    __builtin_amdgcn_s_wait_tensorcnt(0);
  }
#else
  {
    const unsigned int* src = (const unsigned int*)(kn + base);
    unsigned int* dst = (unsigned int*)lds_kn;
    for (int e = tid; e < 4096; e += 128) dst[e] = src[e];
  }
#endif
  __syncthreads();

  int rowtile = quad * 4 + wv;
  int m0 = rowtile * 16;
  const _Float16* qrow = qn + base + (long)(m0 + ln) * 32;
  v16h aq;
#pragma unroll
  for (int v = 0; v < 8; ++v) {
    int k = 2 * v + 16 * half;
    aq[2 * v] = qrow[k];
    aq[2 * v + 1] = qrow[k + 1];
  }

  int wimg = win & 15, wh = wimg >> 2, ww = wimg & 3;
  float s[16][8];
#pragma unroll
  for (int ct = 0; ct < 16; ++ct) {
    v8f acc = {};
    v16h bf;
    const _Float16* kp = &lds_kn[(ct * 16 + ln) * 32 + 16 * half];
#pragma unroll
    for (int v = 0; v < 8; ++v) {
      bf[2 * v] = kp[2 * v];
      bf[2 * v + 1] = kp[2 * v + 1];
    }
    acc = WMMA_F16(aq, bf, acc);
    int j = ct * 16 + ln, rj = j >> 4, cj = j & 15;
    int regj = region_of(wh * 16 + rj) * 3 + region_of(ww * 16 + cj);
#pragma unroll
    for (int r = 0; r < 8; ++r) {
      int i = m0 + r + 8 * half;
      int ri = i >> 4, ci = i & 15;
      int idx = (ri - rj + 15) * 31 + (ci - cj + 15);
      float bv = btbl[idx * 16 + head];
      float sg = 16.0f / (1.0f + __expf(-bv));
      int regi = region_of(wh * 16 + ri) * 3 + region_of(ww * 16 + ci);
      float mk = (regi != regj) ? -100.0f : 0.0f;
      s[ct][r] = acc[r] + sg + mk;
    }
  }
  // row softmax over 256 cols
#pragma unroll
  for (int r = 0; r < 8; ++r) {
    float mx = -3.0e38f;
#pragma unroll
    for (int ct = 0; ct < 16; ++ct) mx = fmaxf(mx, s[ct][r]);
#pragma unroll
    for (int o = 1; o < 16; o <<= 1) mx = fmaxf(mx, __shfl_xor(mx, o, 32));
    float sm = 0.0f;
#pragma unroll
    for (int ct = 0; ct < 16; ++ct) {
      float pp = __expf(s[ct][r] - mx);
      s[ct][r] = pp;
      sm += pp;
    }
#pragma unroll
    for (int o = 1; o < 16; o <<= 1) sm += __shfl_xor(sm, o, 32);
    float inv = 1.0f / sm;
    int mrow = r + 8 * half;
#pragma unroll
    for (int ct = 0; ct < 16; ++ct)
      lds_p[wv][mrow * 256 + ct * 16 + ln] = (_Float16)(s[ct][r] * inv);
  }
  __syncthreads();

  // out(16x32) = P(16x256) * V(256x32)
  v8f o0 = {}, o1 = {};
#pragma unroll
  for (int ks = 0; ks < 8; ++ks) {
    v16h ap;
    const _Float16* pr = &lds_p[wv][ln * 256 + ks * 32];
#pragma unroll
    for (int v = 0; v < 8; ++v) {
      int k = 2 * v + 16 * half;
      ap[2 * v] = pr[k];
      ap[2 * v + 1] = pr[k + 1];
    }
    v16h b0, b1;
    const _Float16* vb = vt + base + (long)(ks * 32 + 16 * half) * 32;
#pragma unroll
    for (int v = 0; v < 8; ++v) {
      b0[2 * v]     = vb[(2 * v) * 32 + ln];
      b0[2 * v + 1] = vb[(2 * v + 1) * 32 + ln];
      b1[2 * v]     = vb[(2 * v) * 32 + 16 + ln];
      b1[2 * v + 1] = vb[(2 * v + 1) * 32 + 16 + ln];
    }
    o0 = WMMA_F16(ap, b0, o0);
    o1 = WMMA_F16(ap, b1, o1);
  }
#pragma unroll
  for (int r = 0; r < 8; ++r) {
    int m = m0 + r + 8 * half;
    long t = (long)win * 256 + m;
    aout[t * 512 + head * 32 + ln]      = (_Float16)o0[r];
    aout[t * 512 + head * 32 + 16 + ln] = (_Float16)o1[r];
  }
}

// ---------------------------------------------------------------------------
// x1 = shortcut + LN(proj_out) ; writes f32 + f16 (window-token order)
// ---------------------------------------------------------------------------
__global__ __launch_bounds__(256) void ln_res1(const float* __restrict__ po,
                                               const float* __restrict__ x,
                                               const float* __restrict__ g,
                                               const float* __restrict__ bta,
                                               float* __restrict__ x1f,
                                               _Float16* __restrict__ x1h) {
  __shared__ float s1[256], s2[256];
  int t = blockIdx.x, tid = threadIdx.x;
  const float* row = po + (long)t * 512;
  float a = row[tid], b = row[tid + 256];
  s1[tid] = a + b;
  s2[tid] = a * a + b * b;
  __syncthreads();
  for (int o = 128; o > 0; o >>= 1) {
    if (tid < o) {
      s1[tid] += s1[tid + o];
      s2[tid] += s2[tid + o];
    }
    __syncthreads();
  }
  float mean = s1[0] * (1.0f / 512.0f);
  float var = s2[0] * (1.0f / 512.0f) - mean * mean;
  float rs = rsqrtf(var + 1e-5f);
  long xoff = canvas_base(t);
  for (int c = tid; c < 512; c += 256) {
    float v = (row[c] - mean) * rs * g[c] + bta[c] + x[xoff + c];
    x1f[(long)t * 512 + c] = v;
    x1h[(long)t * 512 + c] = (_Float16)v;
  }
}

// out = x1 + LN(m) ; scattered back to canvas order (f32)
__global__ __launch_bounds__(256) void ln_res2(const float* __restrict__ mo,
                                               const float* __restrict__ x1f,
                                               const float* __restrict__ g,
                                               const float* __restrict__ bta,
                                               float* __restrict__ out) {
  __shared__ float s1[256], s2[256];
  int t = blockIdx.x, tid = threadIdx.x;
  const float* row = mo + (long)t * 512;
  float a = row[tid], b = row[tid + 256];
  s1[tid] = a + b;
  s2[tid] = a * a + b * b;
  __syncthreads();
  for (int o = 128; o > 0; o >>= 1) {
    if (tid < o) {
      s1[tid] += s1[tid + o];
      s2[tid] += s2[tid + o];
    }
    __syncthreads();
  }
  float mean = s1[0] * (1.0f / 512.0f);
  float var = s2[0] * (1.0f / 512.0f) - mean * mean;
  float rs = rsqrtf(var + 1e-5f);
  long ooff = canvas_base(t);
  for (int c = tid; c < 512; c += 256) {
    float v = (row[c] - mean) * rs * g[c] + bta[c];
    out[ooff + c] = x1f[(long)t * 512 + c] + v;
  }
}

// ---------------------------------------------------------------------------
extern "C" void kernel_launch(void* const* d_in, const int* in_sizes, int n_in,
                              void* d_out, int out_size, void* d_ws, size_t ws_size,
                              hipStream_t stream) {
  const float* x        = (const float*)d_in[0];
  const float* norm1_g  = (const float*)d_in[3];
  const float* norm1_b  = (const float*)d_in[4];
  const float* qkv_w    = (const float*)d_in[5];
  const float* q_bias   = (const float*)d_in[6];
  const float* v_bias   = (const float*)d_in[7];
  const float* logit_sc = (const float*)d_in[8];
  const float* cpb_w1   = (const float*)d_in[9];
  const float* cpb_b1   = (const float*)d_in[10];
  const float* cpb_w2   = (const float*)d_in[11];
  const float* proj_w   = (const float*)d_in[12];
  const float* proj_b   = (const float*)d_in[13];
  const float* norm2_g  = (const float*)d_in[14];
  const float* norm2_b  = (const float*)d_in[15];
  const float* fc1_w    = (const float*)d_in[16];
  const float* fc1_b    = (const float*)d_in[17];
  const float* fc2_w    = (const float*)d_in[18];
  const float* fc2_b    = (const float*)d_in[19];
  (void)in_sizes; (void)n_in; (void)out_size; (void)ws_size;

  // workspace layout (~225 MB, arena-reused)
  char* p = (char*)d_ws;
  size_t o = 0;
  auto take = [&](size_t n) -> char* {
    o = (o + 255) & ~(size_t)255;
    char* r = p + o;
    o += n;
    return r;
  };
  _Float16* wq_h  = (_Float16*)take((size_t)1536 * 512 * 2);
  _Float16* wp_h  = (_Float16*)take((size_t)512 * 512 * 2);
  _Float16* w1_h  = (_Float16*)take((size_t)2048 * 512 * 2);
  _Float16* w2_h  = (_Float16*)take((size_t)512 * 2048 * 2);
  float*    btbl  = (float*)take((size_t)961 * 16 * 4);
  char*     aren1 = take((size_t)16384 * 2048 * 2);      // 67 MB
  char*     aren2 = take((size_t)3 * 16384 * 512 * 4);   // 100.7 MB
  float*    x1f   = (float*)take((size_t)16384 * 512 * 4);
  _Float16* x1h   = (_Float16*)take((size_t)16384 * 512 * 2);

  // arena1 lifetimes: x_win -> qn/kn/v -> proj_out -> hgelu
  _Float16* xwin  = (_Float16*)aren1;
  _Float16* qn_h  = (_Float16*)aren1;
  _Float16* kn_h  = (_Float16*)(aren1 + (size_t)16777216);
  _Float16* vv_h  = (_Float16*)(aren1 + (size_t)33554432);
  float*    projf = (float*)aren1;
  _Float16* hgelu = (_Float16*)aren1;
  // arena2 lifetimes: qkv f32 -> attnout f16 -> mlp f32
  float*    qkvf  = (float*)aren2;
  _Float16* attnh = (_Float16*)aren2;
  float*    mlpf  = (float*)aren2;

  // 1) weight conversion + window gather + CPB table
  cvt_f16<<<(1536 * 512 + 255) / 256, 256, 0, stream>>>(qkv_w, wq_h, 1536 * 512);
  cvt_f16<<<(512 * 512 + 255) / 256, 256, 0, stream>>>(proj_w, wp_h, 512 * 512);
  cvt_f16<<<(2048 * 512 + 255) / 256, 256, 0, stream>>>(fc1_w, w1_h, 2048 * 512);
  cvt_f16<<<(512 * 2048 + 255) / 256, 256, 0, stream>>>(fc2_w, w2_h, 512 * 2048);
  gather_win<<<32768, 256, 0, stream>>>(x, xwin);
  cpb_kernel<<<961, 128, 0, stream>>>(cpb_w1, cpb_b1, cpb_w2, btbl);

  // 2) QKV GEMM (M=16384, N=1536, K=512); block covers 256 N -> grid.y = 6
  gemm_f16<<<dim3(1024, 6), 128, 0, stream>>>(xwin, wq_h, 16384, 1536, 512,
                                              nullptr, 0, nullptr, nullptr,
                                              q_bias, v_bias, qkvf);
  // 3) cosine normalization + logit scale, f32 -> f16
  qkv_norm<<<32768, 256, 0, stream>>>(qkvf, logit_sc, qn_h, kn_h, vv_h);
  // 4) fused windowed attention (64 win x 16 heads x 4 row-quads)
  attn_kernel<<<4096, 128, 0, stream>>>(qn_h, kn_h, vv_h, btbl, attnh);
  // 5) output projection (M=16384, N=512, K=512); grid.y = 2
  gemm_f16<<<dim3(1024, 2), 128, 0, stream>>>(attnh, wp_h, 16384, 512, 512,
                                              proj_b, 1, projf, nullptr,
                                              nullptr, nullptr, nullptr);
  // 6) x1 = shortcut + LN(proj)
  ln_res1<<<16384, 256, 0, stream>>>(projf, x, norm1_g, norm1_b, x1f, x1h);
  // 7) FC1 + exact GELU (M=16384, N=2048, K=512); grid.y = 8
  gemm_f16<<<dim3(1024, 8), 128, 0, stream>>>(x1h, w1_h, 16384, 2048, 512,
                                              fc1_b, 2, nullptr, hgelu,
                                              nullptr, nullptr, nullptr);
  // 8) FC2 (M=16384, N=512, K=2048); grid.y = 2
  gemm_f16<<<dim3(1024, 2), 128, 0, stream>>>(hgelu, w2_h, 16384, 512, 2048,
                                              fc2_b, 1, mlpf, nullptr,
                                              nullptr, nullptr, nullptr);
  // 9) out = x1 + LN(mlp), scattered to canvas order
  ln_res2<<<16384, 256, 0, stream>>>(mlpf, x1f, norm2_g, norm2_b, (float*)d_out);
}
